// Model_11115375362230
// MI455X (gfx1250) — matrix operand order repaired and checked by reading, not compiled, not run on hardware
//
#include <hip/hip_runtime.h>
#include <hip/hip_bf16.h>
#include <math.h>

// ---------------- model dims ----------------
#define BB   8
#define SS   512
#define HH   768
#define LL   12
#define NHH  12
#define HDD  64
#define FFF  3072
#define NCC  6
#define VOC  129            // 2*MAX_REL+1 relative-position vocab

typedef __attribute__((ext_vector_type(16))) __bf16 v16bf;
typedef __attribute__((ext_vector_type(8)))  float  v8f;
typedef __attribute__((ext_vector_type(4)))  __bf16 bf16x4;

// Native hardware convert (v_cvt_pk_bf16_f32 on gfx1250).
__device__ __forceinline__ __bf16 f2bf(float f) { return (__bf16)f; }
__device__ __forceinline__ bf16x4 f4tobf4(float4 f) {
  return (bf16x4){(__bf16)f.x, (__bf16)f.y, (__bf16)f.z, (__bf16)f.w};
}

#define WMMA_BF16(Af, Bf, Cf) \
  __builtin_amdgcn_wmma_f32_16x16x32_bf16(false, (Af), false, (Bf), (short)0, (Cf), false, false)

// A fragment (16x32 bf16): lane<16 row m=lane, K=0..15; lane>=16 K+=16
__device__ __forceinline__ v16bf load_a_frag(const __bf16* base, int stride, int mbase, int kbase, int lane) {
  const int m   = mbase + (lane & 15);
  const int hio = (lane & 16) ? 16 : 0;
  const __bf16* p = base + (size_t)m * stride + kbase + hio;
  v16bf a;
#pragma unroll
  for (int e = 0; e < 8; ++e) a[e]     = p[e];
#pragma unroll
  for (int e = 0; e < 8; ++e) a[8 + e] = p[8 + e];
  return a;
}
// B fragment (32x16 bf16): lane col n=lane&15, K = e + 16*(lane>=16); LDS holds B transposed [n][k]
__device__ __forceinline__ v16bf load_b_frag(const __bf16* base, int stride, int nbase, int kbase, int lane) {
  const int nn  = nbase + (lane & 15);
  const int hio = (lane & 16) ? 16 : 0;
  const __bf16* p = base + (size_t)nn * stride + kbase + hio;
  v16bf b;
#pragma unroll
  for (int e = 0; e < 16; ++e) b[e] = p[e];
  return b;
}

// ---------------- sinusoidal relative-position table [VOC][HDD] ----------------
__global__ void relpos_table_kernel(float* __restrict__ table) {
  int c = blockIdx.x;     // 0..128
  int i = threadIdx.x;    // 0..63
  float pos   = (float)c;
  float expnt = (float)(2 * (i / 2)) / (float)HDD;
  float angle = pos / powf(10000.0f, expnt);
  table[c * HDD + i] = (i & 1) ? cosf(angle) : sinf(angle);
}

// ---------------- embeddings + LayerNorm ----------------
__global__ __launch_bounds__(256) void embed_ln_kernel(
    const int* __restrict__ ids, const float* __restrict__ we, const float* __restrict__ pe,
    const float* __restrict__ te, const float* __restrict__ g, const float* __restrict__ bta,
    float* __restrict__ x) {
  int t = blockIdx.x;                 // token index 0..4095
  int s = t % SS;
  int id = ids[t];
  int tid = threadIdx.x;
  __shared__ float red[256];
  float v[3];
#pragma unroll
  for (int r = 0; r < 3; ++r) {
    int j = tid + r * 256;
    v[r] = we[(size_t)id * HH + j] + pe[(size_t)s * HH + j] + te[j];
  }
  float s0 = v[0] + v[1] + v[2];
  red[tid] = s0; __syncthreads();
  for (int o = 128; o > 0; o >>= 1) { if (tid < o) red[tid] += red[tid + o]; __syncthreads(); }
  float mu = red[0] * (1.0f / HH);
  __syncthreads();
  float d2 = 0.f;
#pragma unroll
  for (int r = 0; r < 3; ++r) { float d = v[r] - mu; d2 += d * d; }
  red[tid] = d2; __syncthreads();
  for (int o = 128; o > 0; o >>= 1) { if (tid < o) red[tid] += red[tid + o]; __syncthreads(); }
  float inv = rsqrtf(red[0] * (1.0f / HH) + 1e-12f);
#pragma unroll
  for (int r = 0; r < 3; ++r) {
    int j = tid + r * 256;
    x[(size_t)t * HH + j] = (v[r] - mu) * inv * g[j] + bta[j];
  }
}

// ---------------- x = LN(x + t) ----------------
__global__ __launch_bounds__(256) void add_ln_kernel(
    float* __restrict__ x, const float* __restrict__ tmp,
    const float* __restrict__ g, const float* __restrict__ bta) {
  int t = blockIdx.x;
  int tid = threadIdx.x;
  __shared__ float red[256];
  float v[3];
#pragma unroll
  for (int r = 0; r < 3; ++r) {
    int j = tid + r * 256;
    v[r] = x[(size_t)t * HH + j] + tmp[(size_t)t * HH + j];
  }
  float s0 = v[0] + v[1] + v[2];
  red[tid] = s0; __syncthreads();
  for (int o = 128; o > 0; o >>= 1) { if (tid < o) red[tid] += red[tid + o]; __syncthreads(); }
  float mu = red[0] * (1.0f / HH);
  __syncthreads();
  float d2 = 0.f;
#pragma unroll
  for (int r = 0; r < 3; ++r) { float d = v[r] - mu; d2 += d * d; }
  red[tid] = d2; __syncthreads();
  for (int o = 128; o > 0; o >>= 1) { if (tid < o) red[tid] += red[tid + o]; __syncthreads(); }
  float inv = rsqrtf(red[0] * (1.0f / HH) + 1e-12f);
#pragma unroll
  for (int r = 0; r < 3; ++r) {
    int j = tid + r * 256;
    x[(size_t)t * HH + j] = (v[r] - mu) * inv * g[j] + bta[j];
  }
}

// ---------------- WMMA bf16 GEMM: C[M,N] = act(A[M,K] @ B[K,N] + bias) ----------------
// fp32 A/B in HBM -> registers (all loads in flight) -> bf16 ping-pong LDS -> v_wmma.
#define LPAD 36   // 18 banks, gcd(18,64)=2 -> 16 fragment rows on distinct banks; 8B aligned
__global__ __launch_bounds__(256) void gemm_bf16_kernel(
    const float* __restrict__ A, const float* __restrict__ B,
    const float* __restrict__ bias, float* __restrict__ C,
    int M, int N, int K, int gelu) {
  __shared__ __bf16 As[2][128][LPAD];  // [buf][m][k]
  __shared__ __bf16 Bs[2][128][LPAD];  // [buf][n][k] (B transposed)
  const int tid  = threadIdx.x;
  const int lane = tid & 31;
  const int wid  = tid >> 5;
  const int bm = blockIdx.y * 128;
  const int bn = blockIdx.x * 128;
  const int waveM = (wid & 3) * 32;   // 4 waves down
  const int waveN = (wid >> 2) * 64;  // 2 waves across
  const int ar = tid >> 3, akc = (tid & 7) * 4;    // A-tile element mapping
  const int bkr = tid >> 5, bnc = (tid & 31) * 4;  // B-tile element mapping

  v8f acc[2][4];
#pragma unroll
  for (int mf = 0; mf < 2; ++mf)
#pragma unroll
    for (int nf = 0; nf < 4; ++nf) acc[mf][nf] = (v8f){0.f,0.f,0.f,0.f,0.f,0.f,0.f,0.f};

  float4 rA[4], rB[4];
  // ---- fetch tile kt into registers (8 independent b128 loads, no interleaved waits)
#define FETCH(kt_)                                                                  \
  {                                                                                 \
    _Pragma("unroll") for (int p = 0; p < 4; ++p)                                   \
      rA[p] = *(const float4*)(A + (size_t)(bm + ar + p * 32) * K + (kt_) + akc);   \
    _Pragma("unroll") for (int p = 0; p < 4; ++p)                                   \
      rB[p] = *(const float4*)(B + (size_t)((kt_) + bkr + p * 8) * N + bn + bnc);   \
  }
  // ---- convert + store registers into LDS buffer `buf_`
#define STAGE(buf_)                                                                 \
  {                                                                                 \
    _Pragma("unroll") for (int p = 0; p < 4; ++p)                                   \
      *(bf16x4*)&As[buf_][ar + p * 32][akc] = f4tobf4(rA[p]);                       \
    _Pragma("unroll") for (int p = 0; p < 4; ++p) {                                 \
      Bs[buf_][bnc][bkr + p * 8]     = f2bf(rB[p].x);                               \
      Bs[buf_][bnc + 1][bkr + p * 8] = f2bf(rB[p].y);                               \
      Bs[buf_][bnc + 2][bkr + p * 8] = f2bf(rB[p].z);                               \
      Bs[buf_][bnc + 3][bkr + p * 8] = f2bf(rB[p].w);                               \
    }                                                                               \
  }

  FETCH(0);
  STAGE(0);
  __syncthreads();

  int cur = 0;
  for (int kt = 0; kt < K; kt += 32) {
    const bool more = (kt + 32) < K;
    if (more) FETCH(kt + 32);          // next tile in flight during this tile's math

    v16bf af[2], bfr[4];
#pragma unroll
    for (int mf = 0; mf < 2; ++mf) af[mf]  = load_a_frag(&As[cur][0][0], LPAD, waveM + mf * 16, 0, lane);
#pragma unroll
    for (int nf = 0; nf < 4; ++nf) bfr[nf] = load_b_frag(&Bs[cur][0][0], LPAD, waveN + nf * 16, 0, lane);
#pragma unroll
    for (int mf = 0; mf < 2; ++mf)
#pragma unroll
      for (int nf = 0; nf < 4; ++nf) acc[mf][nf] = WMMA_BF16(af[mf], bfr[nf], acc[mf][nf]);

    if (more) {
      STAGE(cur ^ 1);                  // safe: buf cur^1 last read before previous barrier
      __syncthreads();                 // single barrier per K-step
      cur ^= 1;
    }
  }

  const int ln  = lane & 15;
  const int hio = (lane & 16) ? 8 : 0;
#pragma unroll
  for (int mf = 0; mf < 2; ++mf)
#pragma unroll
    for (int nf = 0; nf < 4; ++nf) {
      int col = bn + waveN + nf * 16 + ln;
      float bvv = bias[col];
#pragma unroll
      for (int r = 0; r < 8; ++r) {
        int row = bm + waveM + mf * 16 + hio + r;
        float xv = acc[mf][nf][r] + bvv;
        if (gelu) xv = 0.5f * xv * (1.0f + erff(xv * 0.70710678118654752f));
        C[(size_t)row * N + col] = xv;
      }
    }
#undef FETCH
#undef STAGE
}

// ---------------- qrel[b,h,i,c] = q[b,i,h,:] . table[c,:] ----------------
__global__ __launch_bounds__(128) void qrel_kernel(
    const float* __restrict__ q, const float* __restrict__ table, float* __restrict__ qrel) {
  int blk = blockIdx.x;                 // b*NH*S
  int i = blk % SS;
  int h = (blk / SS) % NHH;
  int b = blk / (SS * NHH);
  __shared__ float qr[HDD];
  if (threadIdx.x < HDD) qr[threadIdx.x] = q[((size_t)(b * SS + i)) * HH + h * HDD + threadIdx.x];
  __syncthreads();
  for (int c = threadIdx.x; c < VOC; c += 128) {
    float s = 0.f;
#pragma unroll 8
    for (int d = 0; d < HDD; ++d) s += qr[d] * table[c * HDD + d];
    qrel[(((size_t)(b * NHH + h) * SS) + i) * VOC + c] = s;
  }
}

// ---------------- attention: one wave per (b, h, 16-row i-tile) ----------------
__global__ __launch_bounds__(32) void attn_kernel(
    const float* __restrict__ q, const float* __restrict__ k, const float* __restrict__ v,
    const float* __restrict__ qrel, const float* __restrict__ table,
    const int* __restrict__ mask, float* __restrict__ ctx) {
  const int blk = blockIdx.x;
  const int it = blk % (SS / 16); const int i0 = it * 16;
  const int h  = (blk / (SS / 16)) % NHH;
  const int b  = blk / ((SS / 16) * NHH);
  const int lane = threadIdx.x;

  __shared__ float  sc[16][SS];        // scores -> probs
  __shared__ __bf16 qbf[16][HDD];
  __shared__ __bf16 kbf[16][HDD];
  __shared__ float  arel[16][VOC + 1]; // attention bucketed by clipped distance

  const float scale = 0.125f;          // 1/sqrt(64)

  // q tile: 8 b128 loads per lane -> pk cvts -> b64 LDS stores
#pragma unroll
  for (int p = 0; p < 8; ++p) {
    int f = lane + p * 32;             // float4 index over 16x64 tile
    int m = f >> 4, d4 = (f & 15) * 4;
    float4 qv = *(const float4*)(q + ((size_t)(b * SS + i0 + m)) * HH + h * HDD + d4);
    *(bf16x4*)&qbf[m][d4] = f4tobf4(qv);
  }
  for (int idx = lane; idx < 16 * VOC; idx += 32) arel[idx / VOC][idx % VOC] = 0.f;
  __syncthreads();

  const v16bf a0 = load_a_frag(&qbf[0][0], HDD, 0, 0, lane);
  const v16bf a1 = load_a_frag(&qbf[0][0], HDD, 0, 32, lane);
  const int n   = lane & 15;
  const int hi8 = (lane & 16) ? 8 : 0;

  // ---- scores = (q.kT + qrel_gather) * scale + mask_bias ----
  for (int jt = 0; jt < SS / 16; ++jt) {
    int j0 = jt * 16;
#pragma unroll
    for (int p = 0; p < 8; ++p) {
      int f = lane + p * 32;
      int m = f >> 4, d4 = (f & 15) * 4;
      float4 kv = *(const float4*)(k + ((size_t)(b * SS + j0 + m)) * HH + h * HDD + d4);
      *(bf16x4*)&kbf[m][d4] = f4tobf4(kv);
    }
    __syncthreads();
    v16bf b0 = load_b_frag(&kbf[0][0], HDD, 0, 0, lane);
    v16bf b1 = load_b_frag(&kbf[0][0], HDD, 0, 32, lane);
    v8f c = (v8f){0.f,0.f,0.f,0.f,0.f,0.f,0.f,0.f};
    c = WMMA_BF16(a0, b0, c);
    c = WMMA_BF16(a1, b1, c);
    int j = j0 + n;
    float mb = (1.0f - (float)mask[b * SS + j]) * -10000.0f;
#pragma unroll
    for (int r = 0; r < 8; ++r) {
      int m = r + hi8;
      int i = i0 + m;
      int dist = j - i; dist = dist < -64 ? -64 : (dist > 64 ? 64 : dist);
      float rp = qrel[(((size_t)(b * NHH + h) * SS) + i) * VOC + dist + 64];
      sc[m][j] = (c[r] + rp) * scale + mb;
    }
    __syncthreads();
  }

  // ---- row softmax (2 lanes per row) + distance-bucket accumulation ----
  {
    int row = lane >> 1, half = lane & 1;
    int jb = half * 256;
    float mx = -1e30f;
    for (int j = jb; j < jb + 256; ++j) mx = fmaxf(mx, sc[row][j]);
    mx = fmaxf(mx, __shfl_xor(mx, 1, 32));
    float sum = 0.f;
    for (int j = jb; j < jb + 256; ++j) { float e = __expf(sc[row][j] - mx); sc[row][j] = e; sum += e; }
    sum += __shfl_xor(sum, 1, 32);
    float inv = 1.0f / sum;
    int i = i0 + row;
    for (int j = jb; j < jb + 256; ++j) {
      float p = sc[row][j] * inv; sc[row][j] = p;
      int dist = j - i; dist = dist < -64 ? -64 : (dist > 64 ? 64 : dist);
      atomicAdd(&arel[row][dist + 64], p);
    }
  }
  __syncthreads();

  // ---- ctx = attn @ V (WMMA) + arel @ table (rank-129 rel-value term) ----
  v8f acc[4];
#pragma unroll
  for (int dc = 0; dc < 4; ++dc) acc[dc] = (v8f){0.f,0.f,0.f,0.f,0.f,0.f,0.f,0.f};
  for (int jt = 0; jt < SS / 32; ++jt) {
    int j0 = jt * 32;
    v16bf pa;
    {
      int m   = lane & 15;
      int hio = (lane & 16) ? 16 : 0;
#pragma unroll
      for (int e = 0; e < 8; ++e) pa[e]     = f2bf(sc[m][j0 + hio + e]);
#pragma unroll
      for (int e = 0; e < 8; ++e) pa[8 + e] = f2bf(sc[m][j0 + 8 + hio + e]);
    }
    int hio16 = (lane & 16) ? 16 : 0;
#pragma unroll
    for (int dc = 0; dc < 4; ++dc) {
      int d = dc * 16 + n;
      v16bf vb;
#pragma unroll
      for (int e = 0; e < 16; ++e)
        vb[e] = f2bf(v[((size_t)(b * SS + j0 + hio16 + e)) * HH + h * HDD + d]);
      acc[dc] = WMMA_BF16(pa, vb, acc[dc]);
    }
  }
  float rc[4][8];
#pragma unroll
  for (int dc = 0; dc < 4; ++dc)
#pragma unroll
    for (int r = 0; r < 8; ++r) rc[dc][r] = 0.f;
  for (int cc = 0; cc < VOC; ++cc) {
    float tv[4];
#pragma unroll
    for (int dc = 0; dc < 4; ++dc) tv[dc] = table[cc * HDD + dc * 16 + n];
#pragma unroll
    for (int r = 0; r < 8; ++r) {
      float a = arel[r + hi8][cc];
#pragma unroll
      for (int dc = 0; dc < 4; ++dc) rc[dc][r] += a * tv[dc];
    }
  }
#pragma unroll
  for (int dc = 0; dc < 4; ++dc) {
    int d = dc * 16 + n;
#pragma unroll
    for (int r = 0; r < 8; ++r) {
      int m = r + hi8;
      ctx[((size_t)(b * SS + i0 + m)) * HH + h * HDD + d] = acc[dc][r] + rc[dc][r];
    }
  }
}

// ---------------- pooler / classifier ----------------
__global__ __launch_bounds__(256) void seq_avg_kernel(const float* __restrict__ x, float* __restrict__ avg) {
  int b = blockIdx.x;
  for (int j = threadIdx.x; j < HH; j += 256) {
    float s = 0.f;
    for (int ss = 0; ss < SS; ++ss) s += x[((size_t)(b * SS + ss)) * HH + j];
    avg[b * HH + j] = s * (1.0f / SS);
  }
}
__global__ __launch_bounds__(256) void pooled_kernel(
    const float* __restrict__ x, const float* __restrict__ pw, const float* __restrict__ pb,
    float* __restrict__ pooled) {
  int b = blockIdx.x;
  int nn = blockIdx.y * 256 + threadIdx.x;
  float s = pb[nn];
  for (int kk = 0; kk < HH; ++kk) s += x[((size_t)(b * SS)) * HH + kk] * pw[(size_t)kk * HH + nn];
  pooled[b * HH + nn] = tanhf(s);
}
__global__ void cls_kernel(const float* __restrict__ avg, const float* __restrict__ pooled,
                           const float* __restrict__ cw, const float* __restrict__ cb,
                           float* __restrict__ out) {
  int tid = threadIdx.x;
  if (tid >= BB * NCC) return;
  int b = tid / NCC, c = tid % NCC;
  float s = cb[c];
  for (int j = 0; j < HH; ++j)      s += avg[b * HH + j]    * cw[(size_t)j * NCC + c];
  for (int j = 0; j < HH; ++j)      s += pooled[b * HH + j] * cw[(size_t)(HH + j) * NCC + c];
  out[b * NCC + c] = s;
}

// ---------------- host orchestration ----------------
extern "C" void kernel_launch(void* const* d_in, const int* in_sizes, int n_in,
                              void* d_out, int out_size, void* d_ws, size_t ws_size,
                              hipStream_t stream) {
  const int*   input_ids = (const int*)  d_in[0];
  const int*   attn_mask = (const int*)  d_in[1];
  const float* word_emb  = (const float*)d_in[2];
  const float* pos_emb   = (const float*)d_in[3];
  const float* type_emb  = (const float*)d_in[4];
  const float* ln_emb_g  = (const float*)d_in[5];
  const float* ln_emb_b  = (const float*)d_in[6];
  const float* Wq = (const float*)d_in[7];
  const float* bq = (const float*)d_in[8];
  const float* Wk = (const float*)d_in[9];
  const float* bk = (const float*)d_in[10];
  const float* Wv = (const float*)d_in[11];
  const float* bv = (const float*)d_in[12];
  const float* Wo = (const float*)d_in[13];
  const float* bo = (const float*)d_in[14];
  const float* ln1_g = (const float*)d_in[15];
  const float* ln1_b = (const float*)d_in[16];
  const float* W1 = (const float*)d_in[17];
  const float* b1 = (const float*)d_in[18];
  const float* W2 = (const float*)d_in[19];
  const float* b2 = (const float*)d_in[20];
  const float* ln2_g = (const float*)d_in[21];
  const float* ln2_b = (const float*)d_in[22];
  const float* pool_W = (const float*)d_in[23];
  const float* pool_b = (const float*)d_in[24];
  const float* cls_W  = (const float*)d_in[25];
  const float* cls_b  = (const float*)d_in[26];

  const size_t NT = (size_t)BB * SS;         // 4096 tokens
  float* ws = (float*)d_ws;
  float* x      = ws; ws += NT * HH;
  float* qb     = ws; ws += NT * HH;
  float* kb     = ws; ws += NT * HH;
  float* vb     = ws; ws += NT * HH;
  float* ctxb   = ws; ws += NT * HH;
  float* tmp    = ws; ws += NT * HH;
  float* ffb    = ws; ws += NT * FFF;
  float* qrel   = ws; ws += (size_t)BB * NHH * SS * VOC;
  float* table  = ws; ws += VOC * HDD;
  float* avg    = ws; ws += BB * HH;
  float* pooled = ws; ws += BB * HH;

  relpos_table_kernel<<<VOC, HDD, 0, stream>>>(table);
  embed_ln_kernel<<<(int)NT, 256, 0, stream>>>(input_ids, word_emb, pos_emb, type_emb,
                                               ln_emb_g, ln_emb_b, x);
  const int M = (int)NT;
  const dim3 gH(HH / 128, M / 128);
  const dim3 gF(FFF / 128, M / 128);
  for (int l = 0; l < LL; ++l) {
    const size_t oHH = (size_t)l * HH * HH;
    gemm_bf16_kernel<<<gH, 256, 0, stream>>>(x, Wq + oHH, bq + l * HH, qb, M, HH, HH, 0);
    gemm_bf16_kernel<<<gH, 256, 0, stream>>>(x, Wk + oHH, bk + l * HH, kb, M, HH, HH, 0);
    gemm_bf16_kernel<<<gH, 256, 0, stream>>>(x, Wv + oHH, bv + l * HH, vb, M, HH, HH, 0);
    qrel_kernel<<<BB * NHH * SS, 128, 0, stream>>>(qb, table, qrel);
    attn_kernel<<<BB * NHH * (SS / 16), 32, 0, stream>>>(qb, kb, vb, qrel, table, attn_mask, ctxb);
    gemm_bf16_kernel<<<gH, 256, 0, stream>>>(ctxb, Wo + oHH, bo + l * HH, tmp, M, HH, HH, 0);
    add_ln_kernel<<<(int)NT, 256, 0, stream>>>(x, tmp, ln1_g + l * HH, ln1_b + l * HH);
    gemm_bf16_kernel<<<gF, 256, 0, stream>>>(x, W1 + (size_t)l * HH * FFF, b1 + l * FFF, ffb, M, FFF, HH, 1);
    gemm_bf16_kernel<<<gH, 256, 0, stream>>>(ffb, W2 + (size_t)l * FFF * HH, b2 + l * HH, tmp, M, HH, FFF, 0);
    add_ln_kernel<<<(int)NT, 256, 0, stream>>>(x, tmp, ln2_g + l * HH, ln2_b + l * HH);
  }
  seq_avg_kernel<<<BB, 256, 0, stream>>>(x, avg);
  pooled_kernel<<<dim3(BB, HH / 256), 256, 0, stream>>>(x, pool_W, pool_b, pooled);
  cls_kernel<<<1, 64, 0, stream>>>(avg, pooled, cls_W, cls_b, (float*)d_out);
  (void)in_sizes; (void)n_in; (void)out_size; (void)ws_size;
}
